// Block_89842125898023
// MI455X (gfx1250) — matrix operand order repaired and checked by reading, not compiled, not run on hardware
//
#include <hip/hip_runtime.h>
#include <cstdint>
#include <cstddef>

// ---------------------------------------------------------------------------
// CDNA5 / gfx1250 transformer block: LN1 -> QKV -> causal flash attn -> proj
// (+res) -> LN2 -> MLP (+res).  GEMMs: v_wmma_f32_16x16x32_bf16 with TDM
// (tensor_load_to_lds) tile staging when available.
// ---------------------------------------------------------------------------

typedef __attribute__((ext_vector_type(16))) __bf16 v16bf;
typedef __attribute__((ext_vector_type(8)))  float  v8f;
typedef __attribute__((ext_vector_type(4)))  unsigned int v4u;
typedef __attribute__((ext_vector_type(8)))  int    v8i;
typedef __attribute__((ext_vector_type(4)))  int    v4i;

union AFrag { v16bf v; uint32_t u[8]; };

static __device__ __forceinline__ v8f vzero8() {
    v8f z;
#pragma unroll
    for (int i = 0; i < 8; ++i) z[i] = 0.0f;
    return z;
}

static __device__ __forceinline__ unsigned short f2bf(float f) {
    union { float f; uint32_t u; } c; c.f = f;
    uint32_t u = c.u;
    uint32_t r = u + 0x7FFFu + ((u >> 16) & 1u);  // round-to-nearest-even
    return (unsigned short)(r >> 16);
}

static __device__ __forceinline__ v8f wmma_bf16(const AFrag& a, const AFrag& b, v8f c) {
    return __builtin_amdgcn_wmma_f32_16x16x32_bf16(
        false, a.v, false, b.v, (short)0, c, false, false);
}

// ---------------------------------------------------------------------------
// TDM 2D tile load: global (row-major, 2-byte elements) -> LDS, one D# per
// call, groups mapped per CDNA5 ISA.
// ---------------------------------------------------------------------------
#if __has_builtin(__builtin_amdgcn_tensor_load_to_lds) && \
    __has_builtin(__builtin_amdgcn_s_wait_tensorcnt)
#define USE_TDM 1

static __device__ __forceinline__ uint32_t lds_offset_of(const void* p) {
    return (uint32_t)(size_t)(__attribute__((address_space(3))) const char*)p;
}

static __device__ __forceinline__ void tdm_load_2d(
    const void* gtile, uint32_t lds_off,
    uint32_t td0, uint32_t td1, uint32_t stride0,
    uint32_t tile0, uint32_t tile1) {
    uint64_t ga = (uint64_t)(uintptr_t)gtile;
    // Group0: count=1 | lds_addr | global_addr[56:0] | type=2 at [127:126]
    v4u g0 = { 1u, lds_off, (uint32_t)ga,
               (uint32_t)((ga >> 32) & 0x01FFFFFFu) | 0x80000000u };
    // Group1: data_size=2B; tensor_dim0/1; tile_dim0/1/2; dim0 stride
    v8i g1 = { (int)(1u << 16),                                    // data_size=1 (2B)
               (int)((td0 & 0xFFFFu) << 16),                       // td0[15:0]
               (int)((td0 >> 16) | ((td1 & 0xFFFFu) << 16)),       // td0[31:16] | td1[15:0]
               (int)((td1 >> 16) | (tile0 << 16)),                 // td1[31:16] | tile_dim0
               (int)(tile1),                                       // tile_dim1 | tile_dim2=0
               (int)(stride0),                                     // dim0_stride[31:0]
               0, 0 };
    v4i z = { 0, 0, 0, 0 };
#if __has_include(<hip/amd_detail/amd_gfx1250_TDM.h>)
    v8i z8 = { 0, 0, 0, 0, 0, 0, 0, 0 };
    __builtin_amdgcn_tensor_load_to_lds(g0, g1, z, z, z8, 0);      // 6-arg toolchain
#else
    __builtin_amdgcn_tensor_load_to_lds(g0, g1, z, z, 0);          // 5-arg toolchain
#endif
}
#endif

// ---------------------------------------------------------------------------
// Weight conversion kernels (fp32 -> bf16)
// ---------------------------------------------------------------------------
__global__ __launch_bounds__(256) void k_convert_qkv(
    const float* __restrict__ W, unsigned short* __restrict__ Wb,
    int C, int H, int D) {
    int idx = blockIdx.x * 256 + threadIdx.x;
    int total = H * C * D;
    if (idx >= total) return;
    int h = idx / (C * D);
    int rem = idx - h * C * D;
    int c = rem / D;
    int d = rem - c * D;
    Wb[(size_t)c * (H * D) + h * D + d] = f2bf(W[idx]);
}

__global__ __launch_bounds__(256) void k_convert_plain(
    const float* __restrict__ W, unsigned short* __restrict__ Wb, int n) {
    int idx = blockIdx.x * 256 + threadIdx.x;
    if (idx < n) Wb[idx] = f2bf(W[idx]);
}

// ---------------------------------------------------------------------------
// LayerNorm: one block per row of C=1024, output bf16
// ---------------------------------------------------------------------------
__global__ __launch_bounds__(256) void k_layernorm_bf16(
    const float* __restrict__ x, const float* __restrict__ g,
    const float* __restrict__ be, unsigned short* __restrict__ out, int C) {
    __shared__ float rs[256];
    __shared__ float rq[256];
    int row = blockIdx.x;
    int tid = threadIdx.x;
    const float* xr = x + (size_t)row * C;
    float v[4];
    float s = 0.f, q = 0.f;
#pragma unroll
    for (int j = 0; j < 4; ++j) {
        v[j] = xr[tid + j * 256];
        s += v[j];
        q += v[j] * v[j];
    }
    rs[tid] = s; rq[tid] = q;
    __syncthreads();
    for (int off = 128; off > 0; off >>= 1) {
        if (tid < off) { rs[tid] += rs[tid + off]; rq[tid] += rq[tid + off]; }
        __syncthreads();
    }
    float mean = rs[0] / (float)C;
    float var  = rq[0] / (float)C - mean * mean;
    float rstd = rsqrtf(var + 1e-5f);
    unsigned short* orow = out + (size_t)row * C;
#pragma unroll
    for (int j = 0; j < 4; ++j) {
        int c = tid + j * 256;
        orow[c] = f2bf((v[j] - mean) * rstd * g[c] + be[c]);
    }
}

// ---------------------------------------------------------------------------
// Generic bf16 WMMA GEMM:  Out = epilogue(A[M,K] * B[K,N])
//   block tile 128x64, BK=32, 256 threads (8 waves as 4x2), each wave 32x32.
//   Tiles staged by TDM (tensor_load_to_lds) when available.
//   MAPMODE: 0 = row-major [M,N]; 1 = scatter to [B,H,T,D]; 2 = [B,H,D,T]
// ---------------------------------------------------------------------------
template<int MAPMODE, bool BIAS, bool RELU, bool RES, bool OUTBF16>
__global__ __launch_bounds__(256) void k_gemm_bf16(
    const unsigned short* __restrict__ A,   // [M,K] bf16 row-major
    const unsigned short* __restrict__ Bm,  // [K,N] bf16 row-major
    const float* __restrict__ bias,         // [N] or null
    const float* __restrict__ res,          // [M,N] fp32 or null
    void* __restrict__ Out,
    int M, int N, int K, int T, int H, int D) {
    constexpr int BM = 128, BN = 64, BK = 32;
    __shared__ unsigned short sA[BM * BK];       // [row][k]
    __shared__ unsigned short sB[BK * BN];       // row-major [k][n]

    int tid  = threadIdx.x;
    int wave = tid >> 5, lane = tid & 31;
    int half = lane >> 4, l16 = lane & 15;
    int waveM = wave >> 1, waveN = wave & 1;     // 4 x 2 wave grid
    int bm = blockIdx.y * BM, bn = blockIdx.x * BN;

#if defined(USE_TDM)
    uint32_t ldsA = lds_offset_of(sA);
    uint32_t ldsB = lds_offset_of(sB);
#endif

    v8f acc[2][2];
#pragma unroll
    for (int a = 0; a < 2; ++a)
#pragma unroll
        for (int b = 0; b < 2; ++b) acc[a][b] = vzero8();

    for (int kb = 0; kb < K; kb += BK) {
        __syncthreads();
#if defined(USE_TDM)
        if (wave == 0) {
            tdm_load_2d(A + (size_t)bm * K + kb, ldsA,
                        (uint32_t)K, (uint32_t)M, (uint32_t)K, BK, BM);
            tdm_load_2d(Bm + (size_t)kb * N + bn, ldsB,
                        (uint32_t)N, (uint32_t)K, (uint32_t)N, BN, BK);
        }
        __builtin_amdgcn_s_wait_tensorcnt(0);
#else
#pragma unroll
        for (int r = 0; r < 2; ++r) {
            int idx = tid + r * 256;               // 0..511 -> A 128x32
            int row = idx >> 2;
            int ko  = (idx & 3) * 8;
            *(uint4*)(sA + row * BK + ko) =
                *(const uint4*)(A + (size_t)(bm + row) * K + kb + ko);
        }
        {
            int k  = tid >> 3;                     // B 32x64 row-major
            int n0 = (tid & 7) * 8;
            *(uint4*)(sB + k * BN + n0) =
                *(const uint4*)(Bm + (size_t)(kb + k) * N + bn + n0);
        }
#endif
        // prefetch next tiles toward L2 while WMMAs run
        if (kb + BK < K) {
            __builtin_prefetch(A + (size_t)(bm + (tid & 127)) * K + kb + BK, 0, 0);
            __builtin_prefetch(Bm + (size_t)(kb + BK + (tid & 31)) * N + bn + (tid >> 5) * 8, 0, 0);
        }
        __syncthreads();

        // ---- fragments per ISA layout (CDNA5 ISA 7.12.2) ----
        AFrag af[2], bfm[2];
#pragma unroll
        for (int a = 0; a < 2; ++a) {
            int row = waveM * 32 + a * 16 + l16;
#pragma unroll
            for (int i = 0; i < 8; ++i) {
                int ko = ((i < 4) ? (2 * i) : (16 + 2 * (i - 4))) + half * 8;
                af[a].u[i] = *(const uint32_t*)(sA + row * BK + ko);
            }
        }
#pragma unroll
        for (int b = 0; b < 2; ++b) {
            int col = waveN * 32 + b * 16 + l16;
#pragma unroll
            for (int i = 0; i < 8; ++i) {
                int k = ((i < 4) ? (2 * i) : (16 + 2 * (i - 4))) + half * 8;
                uint32_t lo = sB[k * BN + col];
                uint32_t hi = sB[(k + 1) * BN + col];
                bfm[b].u[i] = lo | (hi << 16);
            }
        }
#pragma unroll
        for (int a = 0; a < 2; ++a)
#pragma unroll
            for (int b = 0; b < 2; ++b)
                acc[a][b] = wmma_bf16(af[a], bfm[b], acc[a][b]);
    }

    // ---- epilogue: C layout VGPR i -> row = i + 8*half, col = l16 ----
#pragma unroll
    for (int a = 0; a < 2; ++a) {
#pragma unroll
        for (int b = 0; b < 2; ++b) {
            int gcol = bn + waveN * 32 + b * 16 + l16;
            float bv = BIAS ? bias[gcol] : 0.0f;
#pragma unroll
            for (int i = 0; i < 8; ++i) {
                int grow = bm + waveM * 32 + a * 16 + i + 8 * half;
                float v = acc[a][b][i] + bv;
                if (RELU) v = fmaxf(v, 0.0f);
                if (RES)  v += res[(size_t)grow * N + gcol];
                if (MAPMODE == 0) {
                    size_t o = (size_t)grow * N + gcol;
                    if (OUTBF16) ((unsigned short*)Out)[o] = f2bf(v);
                    else         ((float*)Out)[o] = v;
                } else {
                    int bb = grow / T, t = grow - bb * T;
                    int h  = gcol / D, d = gcol - h * D;
                    size_t o = (MAPMODE == 1)
                        ? ((size_t)((bb * H + h) * T + t) * D + d)   // [B,H,T,D]
                        : ((size_t)((bb * H + h) * D + d) * T + t);  // [B,H,D,T]
                    ((unsigned short*)Out)[o] = f2bf(v);
                }
            }
        }
    }
}

// ---------------------------------------------------------------------------
// Causal flash attention.  grid = (B*H, T/128), 256 threads = 8 waves.
// Each wave owns 16 query rows; loops over 32-key blocks up to the diagonal.
//   q,k: bf16 [B,H,T,D]   vt: bf16 [B,H,D,T]   out: bf16 [B*T, H*D]
// ---------------------------------------------------------------------------
__global__ __launch_bounds__(256) void k_flash_attn(
    const unsigned short* __restrict__ qg,
    const unsigned short* __restrict__ kg,
    const unsigned short* __restrict__ vtg,
    unsigned short* __restrict__ attn_out,
    int B, int H, int T, int D, float scale) {
    int bh   = blockIdx.x;
    int wave = threadIdx.x >> 5, lane = threadIdx.x & 31;
    int half = lane >> 4, l16 = lane & 15;
    int qbase = blockIdx.y * 128 + wave * 16;

    const unsigned short* q  = qg  + (size_t)bh * T * D;
    const unsigned short* kp = kg  + (size_t)bh * T * D;
    const unsigned short* vt = vtg + (size_t)bh * D * T;

    __shared__ unsigned short sP[8][16 * 32];   // per-wave P staging

    AFrag qa[2];
    {
        int qrow = qbase + l16;
#pragma unroll
        for (int c = 0; c < 2; ++c)
#pragma unroll
            for (int i = 0; i < 8; ++i) {
                int dof = c * 32 + half * 8 + ((i < 4) ? (2 * i) : (16 + 2 * (i - 4)));
                qa[c].u[i] = *(const uint32_t*)(q + (size_t)qrow * D + dof);
            }
    }

    float mrow[8], lrow[8];
#pragma unroll
    for (int i = 0; i < 8; ++i) { mrow[i] = -1e30f; lrow[i] = 0.0f; }
    v8f o[4];
#pragma unroll
    for (int t = 0; t < 4; ++t) o[t] = vzero8();

    int nkb = (qbase + 15) / 32 + 1;            // wave-uniform
    for (int kbi = 0; kbi < nkb; ++kbi) {
        int kb = kbi * 32;
        if (kbi + 1 < nkb) {                    // prefetch next key block
            __builtin_prefetch(kp + (size_t)(kb + 32 + lane) * D, 0, 0);
            __builtin_prefetch(vt + (size_t)(lane * 2) * T + kb + 32, 0, 0);
        }
        // ---- S = Q * K^T ----
        v8f S[2];
#pragma unroll
        for (int kc = 0; kc < 2; ++kc) {
            S[kc] = vzero8();
#pragma unroll
            for (int c = 0; c < 2; ++c) {
                AFrag bf;                        // B-frag: col = key, k = d
                int key = kb + kc * 16 + l16;
#pragma unroll
                for (int i = 0; i < 8; ++i) {
                    int d = c * 32 + half * 8 + ((i < 4) ? (2 * i) : (16 + 2 * (i - 4)));
                    bf.u[i] = *(const uint32_t*)(kp + (size_t)key * D + d);
                }
                S[kc] = wmma_bf16(qa[c], bf, S[kc]);
            }
        }
        // ---- scale + causal mask ----
#pragma unroll
        for (int kc = 0; kc < 2; ++kc) {
            int key = kb + kc * 16 + l16;
#pragma unroll
            for (int i = 0; i < 8; ++i) {
                int qi = qbase + i + 8 * half;
                float v = S[kc][i] * scale;
                S[kc][i] = (key <= qi) ? v : -1e30f;
            }
        }
        // ---- online softmax (row spread over 16 lanes of the half) ----
        float alpha[8];
#pragma unroll
        for (int i = 0; i < 8; ++i) {
            float v = fmaxf(S[0][i], S[1][i]);
#pragma unroll
            for (int m = 1; m < 16; m <<= 1) v = fmaxf(v, __shfl_xor(v, m, 32));
            float mnew = fmaxf(mrow[i], v);
            alpha[i] = __expf(mrow[i] - mnew);
            mrow[i] = mnew;
            float p0 = __expf(S[0][i] - mnew); S[0][i] = p0;
            float p1 = __expf(S[1][i] - mnew); S[1][i] = p1;
            float rs = p0 + p1;
#pragma unroll
            for (int m = 1; m < 16; m <<= 1) rs += __shfl_xor(rs, m, 32);
            lrow[i] = lrow[i] * alpha[i] + rs;
        }
#pragma unroll
        for (int t = 0; t < 4; ++t)
#pragma unroll
            for (int i = 0; i < 8; ++i) o[t][i] *= alpha[i];

        // ---- P: C-layout -> A-fragment via private LDS strip ----
        unsigned short* pw = sP[wave];
#pragma unroll
        for (int kc = 0; kc < 2; ++kc)
#pragma unroll
            for (int i = 0; i < 8; ++i)
                pw[(i + 8 * half) * 32 + kc * 16 + l16] = f2bf(S[kc][i]);
        asm volatile("s_wait_dscnt 0" ::: "memory");
        AFrag pa;
#pragma unroll
        for (int i = 0; i < 8; ++i) {
            int ko = half * 8 + ((i < 4) ? (2 * i) : (16 + 2 * (i - 4)));
            pa.u[i] = *(const uint32_t*)(pw + l16 * 32 + ko);
        }
        // ---- O += P * V  (B-frag from transposed V: pairs contiguous) ----
#pragma unroll
        for (int t = 0; t < 4; ++t) {
            AFrag vf;
            int d = t * 16 + l16;
#pragma unroll
            for (int i = 0; i < 8; ++i) {
                int key = kb + half * 8 + ((i < 4) ? (2 * i) : (16 + 2 * (i - 4)));
                vf.u[i] = *(const uint32_t*)(vt + (size_t)d * T + key);
            }
            o[t] = wmma_bf16(pa, vf, o[t]);
        }
    }

    // ---- normalize + store (concat heads: [B*T, H*D]) ----
    int b = bh / H, h = bh - b * H;
#pragma unroll
    for (int t = 0; t < 4; ++t)
#pragma unroll
        for (int i = 0; i < 8; ++i) {
            int qrow = qbase + i + 8 * half;
            float v = o[t][i] / lrow[i];
            attn_out[(size_t)(b * T + qrow) * (H * D) + h * D + t * 16 + l16] = f2bf(v);
        }
}

// ---------------------------------------------------------------------------
// Host orchestration
// ---------------------------------------------------------------------------
extern "C" void kernel_launch(void* const* d_in, const int* in_sizes, int n_in,
                              void* d_out, int out_size, void* d_ws, size_t ws_size,
                              hipStream_t stream) {
    (void)in_sizes; (void)n_in; (void)out_size; (void)ws_size;
    const int B = 4, T = 2048, C = 1024, H = 16, D = 64;
    const int M = B * T;           // 8192
    const int F = 4 * C;           // 4096

    const float* x     = (const float*)d_in[0];
    const float* Wq    = (const float*)d_in[1];
    const float* Wk    = (const float*)d_in[2];
    const float* Wv    = (const float*)d_in[3];
    const float* Wp    = (const float*)d_in[4];
    const float* bp    = (const float*)d_in[5];
    const float* ln1_g = (const float*)d_in[6];
    const float* ln1_b = (const float*)d_in[7];
    const float* ln2_g = (const float*)d_in[8];
    const float* ln2_b = (const float*)d_in[9];
    const float* W1    = (const float*)d_in[10];
    const float* b1    = (const float*)d_in[11];
    const float* W2    = (const float*)d_in[12];
    const float* b2    = (const float*)d_in[13];
    float* out = (float*)d_out;

    size_t off = 0;
    char* ws = (char*)d_ws;
    auto take = [&](size_t bytes) -> void* {
        void* p = ws + off;
        off += (bytes + 255) & ~(size_t)255;
        return p;
    };
    unsigned short* wq_b = (unsigned short*)take((size_t)C * C * 2);
    unsigned short* wk_b = (unsigned short*)take((size_t)C * C * 2);
    unsigned short* wv_b = (unsigned short*)take((size_t)C * C * 2);
    unsigned short* wp_b = (unsigned short*)take((size_t)C * C * 2);
    unsigned short* w1_b = (unsigned short*)take((size_t)C * F * 2);
    unsigned short* w2_b = (unsigned short*)take((size_t)F * C * 2);
    unsigned short* h1   = (unsigned short*)take((size_t)M * C * 2);
    unsigned short* qb   = (unsigned short*)take((size_t)M * C * 2);
    unsigned short* kb   = (unsigned short*)take((size_t)M * C * 2);
    unsigned short* vtb  = (unsigned short*)take((size_t)M * C * 2);
    unsigned short* atb  = (unsigned short*)take((size_t)M * C * 2);
    float*          x2   = (float*)take((size_t)M * C * 4);
    unsigned short* h2   = (unsigned short*)take((size_t)M * C * 2);
    unsigned short* ff1  = (unsigned short*)take((size_t)M * F * 2);

    dim3 blk(256);
    k_convert_qkv<<<(H * C * D) / 256, blk, 0, stream>>>(Wq, wq_b, C, H, D);
    k_convert_qkv<<<(H * C * D) / 256, blk, 0, stream>>>(Wk, wk_b, C, H, D);
    k_convert_qkv<<<(H * C * D) / 256, blk, 0, stream>>>(Wv, wv_b, C, H, D);
    k_convert_plain<<<(C * C) / 256, blk, 0, stream>>>(Wp, wp_b, C * C);
    k_convert_plain<<<(C * F) / 256, blk, 0, stream>>>(W1, w1_b, C * F);
    k_convert_plain<<<(F * C) / 256, blk, 0, stream>>>(W2, w2_b, F * C);

    k_layernorm_bf16<<<M, blk, 0, stream>>>(x, ln1_g, ln1_b, h1, C);

    dim3 g1(C / 64, M / 128);
    k_gemm_bf16<1, false, false, false, true><<<g1, blk, 0, stream>>>(
        h1, wq_b, nullptr, nullptr, qb, M, C, C, T, H, D);
    k_gemm_bf16<1, false, false, false, true><<<g1, blk, 0, stream>>>(
        h1, wk_b, nullptr, nullptr, kb, M, C, C, T, H, D);
    k_gemm_bf16<2, false, false, false, true><<<g1, blk, 0, stream>>>(
        h1, wv_b, nullptr, nullptr, vtb, M, C, C, T, H, D);

    k_flash_attn<<<dim3(B * H, T / 128), blk, 0, stream>>>(
        qb, kb, vtb, atb, B, H, T, D, 0.03125f);

    k_gemm_bf16<0, true, false, true, false><<<g1, blk, 0, stream>>>(
        atb, wp_b, bp, x, x2, M, C, C, T, H, D);

    k_layernorm_bf16<<<M, blk, 0, stream>>>(x2, ln2_g, ln2_b, h2, C);

    dim3 g2(F / 64, M / 128);
    k_gemm_bf16<0, true, true, false, true><<<g2, blk, 0, stream>>>(
        h2, w1_b, b1, nullptr, ff1, M, F, C, T, H, D);
    k_gemm_bf16<0, true, false, true, false><<<g1, blk, 0, stream>>>(
        ff1, w2_b, b2, x2, out, M, C, F, T, H, D);
}
